// MultiHeadAttention_43104291783119
// MI455X (gfx1250) — matrix operand order repaired and checked by reading, not compiled, not run on hardware
//
#include <hip/hip_runtime.h>

typedef _Float16 f16;
typedef __attribute__((ext_vector_type(8)))  _Float16 v8h;
typedef __attribute__((ext_vector_type(16))) _Float16 v16h;
typedef __attribute__((ext_vector_type(8)))  float    v8f;
typedef __attribute__((ext_vector_type(4)))  unsigned int u32x4;
typedef __attribute__((ext_vector_type(8)))  int      i32x8;
typedef __attribute__((ext_vector_type(4)))  int      i32x4;

namespace {
constexpr int Bc = 2;      // batch
constexpr int Mc = 2048;   // keys/values
constexpr int Nc = 2048;   // queries
constexpr int KH = 16;     // heads
constexpr int Pc = 1024;   // value-in dim
constexpr int Qc = 1024;   // out dim
constexpr int Dc = 64;     // head dim
constexpr int KD = KH * Dc; // 1024
}

__device__ __forceinline__ v16h frag_from(const f16* p0, const f16* p1) {
  v8h lo = *(const v8h*)p0;
  v8h hi = *(const v8h*)p1;
  return __builtin_shufflevector(lo, hi, 0,1,2,3,4,5,6,7,8,9,10,11,12,13,14,15);
}

__device__ __forceinline__ v8f wmma16(v16h a, v16h b, v8f c) {
  return __builtin_amdgcn_wmma_f32_16x16x32_f16(false, a, false, b, (short)0, c,
                                                false, false);
}

// ---------------------------------------------------------------------------
// Tensor Data Mover: DMA a 2D f16 tile (dim0 elems x dim1 rows, row stride
// stride0 elems) from global memory into LDS at lds_byte_addr.
// D# layout per CDNA5 ISA:
//   group0: [1:0]=count=1, [63:32]=lds_addr, [120:64]=global_addr,
//           [127:126]=type=2
//   group1: [17:16]=data_size(1 => 2B), [79:48]=tensor_dim0,
//           [111:80]=tensor_dim1, [127:112]=tile_dim0, [143:128]=tile_dim1,
//           [207:160]=tensor_dim0_stride
// This toolchain exposes the 6-arg builtin (extra int32x8 group before cpol).
// ---------------------------------------------------------------------------
__device__ __forceinline__ void tdm_load_2d_f16(unsigned int lds_byte_addr,
                                                const void* gptr,
                                                unsigned int dim0,
                                                unsigned int dim1,
                                                unsigned int stride0) {
  const unsigned long long ga = (unsigned long long)(uintptr_t)gptr;
  u32x4 g0 = { 1u,                                   // count=1 user descriptor
               lds_byte_addr,                        // LDS destination (bytes)
               (unsigned int)ga,                     // global addr [95:64]
               (unsigned int)((ga >> 32) & 0x01FFFFFFu) | (2u << 30) }; // type=2
  i32x8 g1 = { (int)(1u << 16),                      // wg_mask=0, data_size=2B
               (int)((dim0 & 0xFFFFu) << 16),        // tensor_dim0 lo16
               (int)(((dim0 >> 16) & 0xFFFFu) |      // tensor_dim0 hi16
                     ((dim1 & 0xFFFFu) << 16)),      // tensor_dim1 lo16
               (int)(((dim1 >> 16) & 0xFFFFu) |      // tensor_dim1 hi16
                     ((dim0 & 0xFFFFu) << 16)),      // tile_dim0
               (int)(dim1 & 0xFFFFu),                // tile_dim1 (tile_dim2=0)
               (int)stride0,                         // tensor_dim0_stride lo32
               0, 0 };
  i32x4 gz4 = { 0, 0, 0, 0 };
  i32x8 gz8 = { 0, 0, 0, 0, 0, 0, 0, 0 };
  __builtin_amdgcn_tensor_load_to_lds(g0, g1, gz4, gz4, gz8, 0);
}

// ---------------------------------------------------------------------------
// Projection GEMM: Y[b,k,row,d] = sum_p X[b,row,p] * W[k,p,d] (+ bias[k,d])
// grid = (rows/128, B*KH), block = 256 (8 waves; each wave 16 rows x 64 cols)
// ---------------------------------------------------------------------------
__global__ __launch_bounds__(256) void mha_proj_gemm(
    const float* __restrict__ X, const float* __restrict__ W,
    const float* __restrict__ bias, f16* __restrict__ Yo,
    int rows, int inner) {
  __shared__ __align__(16) f16 sX[128][32];
  __shared__ __align__(16) f16 sWT[64][32];
  const int tid  = threadIdx.x;
  const int w    = tid >> 5, lane = tid & 31;
  const int g    = lane >> 4, col = lane & 15;
  const int b    = blockIdx.y >> 4, k = blockIdx.y & 15;
  const int r0   = blockIdx.x * 128;
  const float* Xb = X + (size_t)b * rows * inner;
  const float* Wb = W + (size_t)k * inner * Dc;
  f16* Ob = Yo + ((size_t)blockIdx.y * rows + r0) * Dc;

  const v8f vzero = {0.f,0.f,0.f,0.f,0.f,0.f,0.f,0.f};
  v8f acc[4];
  for (int t = 0; t < 4; ++t) acc[t] = vzero;

  const int xr = tid >> 1, xc = (tid & 1) * 16;        // X-tile loader coords
  const int wp = tid & 31, wd = (tid >> 5) * 8;        // W-tile loader coords
  const int arow = w * 16 + col;                       // A-fragment row

  for (int pc = 0; pc < inner; pc += 32) {
    __syncthreads();
    const float* xg = Xb + (size_t)(r0 + xr) * inner + pc + xc;
    #pragma unroll
    for (int i = 0; i < 16; ++i) sX[xr][xc + i] = (f16)xg[i];
    const float* wg = Wb + (size_t)(pc + wp) * Dc + wd;
    #pragma unroll
    for (int i = 0; i < 8; ++i) sWT[wd + i][wp] = (f16)wg[i];
    __syncthreads();

    v16h a = frag_from(&sX[arow][8*g], &sX[arow][8*g + 16]);
    #pragma unroll
    for (int t = 0; t < 4; ++t) {
      v16h bt = frag_from(&sWT[16*t + col][8*g], &sWT[16*t + col][8*g + 16]);
      acc[t] = wmma16(a, bt, acc[t]);
    }
  }

  #pragma unroll
  for (int t = 0; t < 4; ++t) {
    const int d = 16*t + col;
    const float bv = bias ? bias[(size_t)k * Dc + d] : 0.f;
    #pragma unroll
    for (int r = 0; r < 8; ++r) {
      const int rr = w*16 + 8*g + r;
      Ob[(size_t)rr * Dc + d] = (f16)(acc[t][r] + bv);
    }
  }
}

// ---------------------------------------------------------------------------
// Flash attention over M (softmax axis). grid = (N/128, B*KH), block = 256.
// Each wave: 16 queries x full D=64. m-tiles of 32. ctx[b,n,k*64+d] (f16).
// K tile (contiguous 4KB) arrives via the Tensor Data Mover.
// ---------------------------------------------------------------------------
__global__ __launch_bounds__(256) void mha_flash_attn(
    const f16* __restrict__ Qh, const f16* __restrict__ Kh,
    const f16* __restrict__ Vh, const float* __restrict__ mask,
    f16* __restrict__ Ctx) {
  __shared__ __align__(16) f16 sK[32][64];
  __shared__ __align__(16) f16 sVT[64][32];
  __shared__ __align__(16) f16 sP[8][16][32];
  const int tid  = threadIdx.x;
  const int w    = tid >> 5, lane = tid & 31;
  const int g    = lane >> 4, col = lane & 15;
  const int bk   = blockIdx.y;
  const int b    = bk >> 4, k = bk & 15;
  const int nrow = blockIdx.x * 128 + w * 16;
  const unsigned int ldsK = (unsigned int)(uintptr_t)&sK[0][0];

  // Q A-fragments for the wave's 16 rows (two K=32 chunks over D=64)
  const f16* qg = Qh + ((size_t)bk * Nc + nrow + col) * Dc;
  v16h aq0 = frag_from(qg + 8*g,      qg + 8*g + 16);
  v16h aq1 = frag_from(qg + 32 + 8*g, qg + 32 + 8*g + 16);

  const v8f vzero = {0.f,0.f,0.f,0.f,0.f,0.f,0.f,0.f};
  v8f acc[4];
  for (int t = 0; t < 4; ++t) acc[t] = vzero;
  float rm[8], rs[8];
  #pragma unroll
  for (int r = 0; r < 8; ++r) { rm[r] = -3.0e38f; rs[r] = 0.f; }

  const int vr = tid & 31, vd = (tid >> 5) * 8;  // V-tile loader coords

  for (int m0 = 0; m0 < Mc; m0 += 32) {
    __syncthreads();
    // K tile 32x64 f16 is contiguous in memory: one TDM descriptor from wave 0
    if (w == 0) {
      tdm_load_2d_f16(ldsK, Kh + ((size_t)bk * Mc + m0) * Dc,
                      32u * 64u, 1u, 32u * 64u);
    }
    // V tile transposed into LDS by all threads (needs layout change)
    const f16* vg = Vh + ((size_t)bk * Mc + m0 + vr) * Dc + vd;
    #pragma unroll
    for (int i = 0; i < 8; ++i) sVT[vd + i][vr] = vg[i];
    // prefetch next iteration's K/V tiles while this one is consumed
    if (m0 + 32 < Mc) {
      __builtin_prefetch(Kh + ((size_t)bk * Mc + m0 + 32) * Dc, 0, 3);
      __builtin_prefetch(Vh + ((size_t)bk * Mc + m0 + 32) * Dc, 0, 3);
    }
    if (w == 0) __builtin_amdgcn_s_wait_tensorcnt(0);
    __syncthreads();

    // S = Q * K^T : two 16x16 tiles (m columns), K-dim = 64 in two chunks
    v16h b00 = frag_from(&sK[col][8*g],        &sK[col][8*g + 16]);
    v16h b01 = frag_from(&sK[col][32 + 8*g],   &sK[col][32 + 8*g + 16]);
    v16h b10 = frag_from(&sK[16+col][8*g],     &sK[16+col][8*g + 16]);
    v16h b11 = frag_from(&sK[16+col][32 + 8*g],&sK[16+col][32 + 8*g + 16]);
    v8f s0 = wmma16(aq0, b00, vzero);
    s0     = wmma16(aq1, b01, s0);
    v8f s1 = wmma16(aq0, b10, vzero);
    s1     = wmma16(aq1, b11, s1);

    // online softmax over m (rows = queries; reduce across 16-lane groups)
    #pragma unroll
    for (int r = 0; r < 8; ++r) {
      const int n = nrow + 8*g + r;
      float x0 = (s0[r] + mask[(size_t)(m0 + col) * Nc + n]) * 0.125f;
      float x1 = (s1[r] + mask[(size_t)(m0 + 16 + col) * Nc + n]) * 0.125f;
      float mx = fmaxf(x0, x1);
      mx = fmaxf(mx, __shfl_xor(mx, 1, 16));
      mx = fmaxf(mx, __shfl_xor(mx, 2, 16));
      mx = fmaxf(mx, __shfl_xor(mx, 4, 16));
      mx = fmaxf(mx, __shfl_xor(mx, 8, 16));
      const float nm = fmaxf(rm[r], mx);
      const float sc = __expf(rm[r] - nm);
      const float p0 = __expf(x0 - nm);
      const float p1 = __expf(x1 - nm);
      float ps = p0 + p1;
      ps += __shfl_xor(ps, 1, 16);
      ps += __shfl_xor(ps, 2, 16);
      ps += __shfl_xor(ps, 4, 16);
      ps += __shfl_xor(ps, 8, 16);
      rs[r] = rs[r] * sc + ps;
      rm[r] = nm;
      #pragma unroll
      for (int t = 0; t < 4; ++t) acc[t][r] *= sc;
      sP[w][8*g + r][col]      = (f16)p0;
      sP[w][8*g + r][16 + col] = (f16)p1;
    }
    // per-wave LDS round-trip: C-layout -> A-fragment layout
    asm volatile("s_wait_dscnt 0x0" ::: "memory");
    v16h ap = frag_from(&sP[w][col][8*g], &sP[w][col][8*g + 16]);
    #pragma unroll
    for (int t = 0; t < 4; ++t) {
      v16h bvf = frag_from(&sVT[16*t + col][8*g], &sVT[16*t + col][8*g + 16]);
      acc[t] = wmma16(ap, bvf, acc[t]);
    }
  }

  #pragma unroll
  for (int t = 0; t < 4; ++t) {
    const int d = 16*t + col;
    #pragma unroll
    for (int r = 0; r < 8; ++r) {
      const int n = nrow + 8*g + r;
      Ctx[((size_t)b * Nc + n) * KD + k * Dc + d] = (f16)(acc[t][r] / rs[r]);
    }
  }
}

// ---------------------------------------------------------------------------
// Output projection: out[b,n,q] = sum_{k,d} ctx[b,n,k*64+d]*Ty[k,q,d] + To1[q]
// grid = (N/128, Q/64, B), block = 256. ctx tile arrives via TDM (2D strided).
// ---------------------------------------------------------------------------
__global__ __launch_bounds__(256) void mha_out_gemm(
    const f16* __restrict__ Ctx, const float* __restrict__ Ty,
    const float* __restrict__ To1, float* __restrict__ Out) {
  __shared__ __align__(16) f16 sA[128][32];
  __shared__ __align__(16) f16 sWT[64][32];
  const int tid  = threadIdx.x;
  const int w    = tid >> 5, lane = tid & 31;
  const int g    = lane >> 4, col = lane & 15;
  const int b    = blockIdx.z;
  const int n0   = blockIdx.x * 128;
  const int q0   = blockIdx.y * 64;
  const unsigned int ldsA = (unsigned int)(uintptr_t)&sA[0][0];

  const v8f vzero = {0.f,0.f,0.f,0.f,0.f,0.f,0.f,0.f};
  v8f acc[4];
  for (int t = 0; t < 4; ++t) acc[t] = vzero;

  const int wq = tid >> 2, wd = (tid & 3) * 8;    // W-tile loader coords
  const int arow = w * 16 + col;

  for (int kc = 0; kc < KD; kc += 32) {
    __syncthreads();
    // ctx tile: 128 rows x 32 f16, row stride KD elems -> one 2D TDM descriptor
    if (w == 0) {
      tdm_load_2d_f16(ldsA, Ctx + ((size_t)b * Nc + n0) * KD + kc,
                      32u, 128u, (unsigned int)KD);
    }
    // W2[(k,d), q] = Ty[k, q, d]; this 32-wide kd chunk has constant k
    const int kk = kc >> 6;
    const int d0 = kc & 63;
    const float* wg = Ty + ((size_t)kk * Qc + q0 + wq) * Dc + d0 + wd;
    #pragma unroll
    for (int i = 0; i < 8; ++i) sWT[wq][wd + i] = (f16)wg[i];
    if (w == 0) __builtin_amdgcn_s_wait_tensorcnt(0);
    __syncthreads();

    v16h a = frag_from(&sA[arow][8*g], &sA[arow][8*g + 16]);
    #pragma unroll
    for (int t = 0; t < 4; ++t) {
      v16h bt = frag_from(&sWT[16*t + col][8*g], &sWT[16*t + col][8*g + 16]);
      acc[t] = wmma16(a, bt, acc[t]);
    }
  }

  #pragma unroll
  for (int t = 0; t < 4; ++t) {
    const int q = q0 + 16*t + col;
    const float bv = To1[q];
    #pragma unroll
    for (int r = 0; r < 8; ++r) {
      const int n = n0 + w*16 + 8*g + r;
      Out[((size_t)b * Nc + n) * Qc + q] = acc[t][r] + bv;
    }
  }
}

// ---------------------------------------------------------------------------
extern "C" void kernel_launch(void* const* d_in, const int* in_sizes, int n_in,
                              void* d_out, int out_size, void* d_ws, size_t ws_size,
                              hipStream_t stream) {
  const float* y_q  = (const float*)d_in[0];  // [B,N,Qp]
  const float* x_kv = (const float*)d_in[1];  // [B,M,Pp]
  const float* mask = (const float*)d_in[2];  // [M,N]
  const float* Lx   = (const float*)d_in[3];  // [K,Pp,Dp]
  const float* Ly   = (const float*)d_in[4];  // [K,Qp,Dp]
  const float* Lo   = (const float*)d_in[5];  // [1,K,1,Dp]
  const float* Tx   = (const float*)d_in[6];  // [K,P,D]
  const float* Ty   = (const float*)d_in[7];  // [K,Q,D]
  const float* To0  = (const float*)d_in[8];  // [1,K,1,D]
  const float* To1  = (const float*)d_in[9];  // [1,1,Q]
  float* out = (float*)d_out;

  const size_t headElems = (size_t)Bc * KH * Nc * Dc; // 4,194,304 (M == N)
  f16* qh  = (f16*)d_ws;
  f16* kh  = qh + headElems;
  f16* vh  = kh + headElems;
  f16* ctx = vh + headElems;                          // [B, N, K*D]

  dim3 blk(256);
  // projections: q = y_q@Ly + Lo ; kproj = x_kv@Lx ; vproj = x_kv@Tx + To0
  mha_proj_gemm<<<dim3(Nc / 128, Bc * KH), blk, 0, stream>>>(
      y_q, Ly, Lo, qh, Nc, Qc);
  mha_proj_gemm<<<dim3(Mc / 128, Bc * KH), blk, 0, stream>>>(
      x_kv, Lx, nullptr, kh, Mc, Pc);
  mha_proj_gemm<<<dim3(Mc / 128, Bc * KH), blk, 0, stream>>>(
      x_kv, Tx, To0, vh, Mc, Pc);
  // fused attention (softmax over M) -> ctx
  mha_flash_attn<<<dim3(Nc / 128, Bc * KH), blk, 0, stream>>>(
      qh, kh, vh, mask, ctx);
  // folded output projection
  mha_out_gemm<<<dim3(Nc / 128, Qc / 64, Bc), blk, 0, stream>>>(
      ctx, Ty, To1, out);
}
